// SLRDifferentialModule_9234179686594
// MI455X (gfx1250) — matrix operand order repaired and checked by reading, not compiled, not run on hardware
//
#include <hip/hip_runtime.h>

typedef __attribute__((ext_vector_type(16))) __bf16 v16bf;
typedef __attribute__((ext_vector_type(8)))  __bf16 v8bf;
typedef __attribute__((ext_vector_type(4)))  __bf16 v4bf;
typedef __attribute__((ext_vector_type(8)))  float  v8f;
typedef __attribute__((ext_vector_type(4)))  float  v4f;

namespace {
constexpr int BATCH = 16;
constexpr int KQ    = 2048;   // query length
constexpr int WKEY  = 2048;   // key/value length
constexpr int DIM   = 128;

constexpr int WC  = 64;       // keys per streamed chunk
constexpr int QT  = 16;       // query rows per wave (WMMA M)
constexpr int NQW = 8;        // query tiles per block
constexpr int NWV = 16;       // waves per block: 8 tiles x 2 branches
constexpr int BQ  = QT * NQW; // 128 query rows per block

// bf16 LDS strides; rows 32B-aligned so v16bf loads lower to 2x ds_load_b128
constexpr int KSTR = DIM + 16; // 144 bf16 -> 288B rows
constexpr int VSTR = WC + 16;  // 80 bf16  -> 160B rows
constexpr int PSTR = WC + 8;   // 72 bf16  -> 144B rows (16B aligned)
constexpr int CSTR = 68;       // combine buffer row stride (floats)
}

union __align__(32) SharedMem {
  struct {
    __bf16 k1[WC][KSTR];        // 18432 B
    __bf16 k2[WC][KSTR];        // 18432 B
    __bf16 vt[DIM][VSTR];       // 20480 B (V transposed: vt[d][w])
    __bf16 p[NWV][QT][PSTR];    // 36864 B (per-wave P re-layout bounce)
  } s;                          // ~92 KB (of 320 KB/WGP)
  float combine[NQW][QT][CSTR]; // 34.8 KB, reused after main loop
};

__device__ __forceinline__ v8f wmma_bf16(v16bf a, v16bf b, v8f c) {
  // (neg_a, A, neg_b, B, c_mod, C, reuse_a, reuse_b)
  return __builtin_amdgcn_wmma_f32_16x16x32_bf16(false, a, false, b, (short)0, c,
                                                 false, false);
}

// One butterfly step of a 16-lane max reduction, entirely on the VALU via DPP16.
template <int CTRL>
__device__ __forceinline__ float dpp_max_step(float x) {
  int xi = __builtin_bit_cast(int, x);
  int yi = __builtin_amdgcn_update_dpp(xi, xi, CTRL, 0xF, 0xF, false);
  return fmaxf(x, __builtin_bit_cast(float, yi));
}

__device__ __forceinline__ float rowmax16(float t) {
  t = dpp_max_step<0xB1>(t);   // quad_perm(1,0,3,2): xor 1
  t = dpp_max_step<0x4E>(t);   // quad_perm(2,3,0,1): xor 2
  t = dpp_max_step<0x141>(t);  // row_half_mirror: combine quads within 8
  t = dpp_max_step<0x140>(t);  // row_mirror: combine halves within 16
  return t;
}

__global__ void __launch_bounds__(512)
diff_attn_kernel(const float* __restrict__ Q1, const float* __restrict__ Q2,
                 const float* __restrict__ K1, const float* __restrict__ K2,
                 const float* __restrict__ V,  const float* __restrict__ lamLogit,
                 float* __restrict__ out) {
  __shared__ SharedMem sh;

  const int tid    = threadIdx.x;
  const int wave   = tid >> 5;
  const int lane   = tid & 31;
  const int h      = lane >> 4;    // lane half (WMMA fragment layouts)
  const int ln     = lane & 15;
  const int branch = wave >> 3;    // 0: (Q1,K1)   1: (Q2,K2)
  const int wv     = wave & 7;     // query-tile index within block

  const int b  = blockIdx.y;
  const int q0 = blockIdx.x * BQ + wv * QT;

  // log2(e)/sqrt(128): scores land in log2 domain -> bare v_exp_f32 for softmax
  const float scale = 0.08838834764831845f * 1.4426950408889634f;

  // ---- per-branch Q tile as A-fragments (16x32 bf16 per d-chunk), pre-scaled ----
  const float* Qg = branch ? Q2 : Q1;
  v16bf aQ[4];
  {
    const float* qp = Qg + ((size_t)b * KQ + (q0 + ln)) * DIM;
#pragma unroll
    for (int kc = 0; kc < 4; ++kc) {
#pragma unroll
      for (int i = 0; i < 8; ++i) {
        aQ[kc][i]     = (__bf16)(qp[kc * 32 + 16 * h +     i] * scale);
        aQ[kc][8 + i] = (__bf16)(qp[kc * 32 + 16 * h + 8 + i] * scale);
      }
    }
  }

  v16bf ones;                      // B fragment of all-ones: row sums via WMMA
#pragma unroll
  for (int i = 0; i < 16; ++i) ones[i] = (__bf16)1.0f;

  v8f O[8];                        // 16x128 fp32 output accumulator
  v8f Lacc;                        // per-row softmax denominator (all lanes of a row equal)
  float rm[8];                     // running row max (log2 domain)
#pragma unroll
  for (int nt = 0; nt < 8; ++nt)
#pragma unroll
    for (int r = 0; r < 8; ++r) O[nt][r] = 0.f;
#pragma unroll
  for (int r = 0; r < 8; ++r) { Lacc[r] = 0.f; rm[r] = -INFINITY; }

  for (int w0 = 0; w0 < WKEY; w0 += WC) {
    // ---- stage K1,K2,V chunk (64x128 fp32 each); convert fp32->bf16 ONCE ----
    const size_t gbase = ((size_t)b * WKEY + w0) * DIM;
#pragma unroll
    for (int it = 0; it < 4; ++it) {
      int idx = tid + 512 * it;       // 2048 float4 per tensor per chunk
      int row = idx >> 5;
      int c4  = idx & 31;
      v4f k1v = *(const v4f*)(K1 + gbase + (size_t)row * DIM + c4 * 4);
      v4f k2v = *(const v4f*)(K2 + gbase + (size_t)row * DIM + c4 * 4);
      v4f vv  = *(const v4f*)(V  + gbase + (size_t)row * DIM + c4 * 4);
      v4bf k1b, k2b;
#pragma unroll
      for (int j = 0; j < 4; ++j) { k1b[j] = (__bf16)k1v[j]; k2b[j] = (__bf16)k2v[j]; }
      *(v4bf*)&sh.s.k1[row][c4 * 4] = k1b;
      *(v4bf*)&sh.s.k2[row][c4 * 4] = k2b;
#pragma unroll
      for (int j = 0; j < 4; ++j) sh.s.vt[c4 * 4 + j][row] = (__bf16)vv[j];
    }
    if (w0 + WC < WKEY) {            // pull next chunk toward the caches
      const size_t nb = gbase + (size_t)WC * DIM;
      __builtin_prefetch(K1 + nb + tid * 16, 0, 0);
      __builtin_prefetch(K2 + nb + tid * 16, 0, 0);
      __builtin_prefetch(V  + nb + tid * 16, 0, 0);
    }
    __syncthreads();

    const __bf16 (*sK)[KSTR] = branch ? sh.s.k2 : sh.s.k1;

    // ---- S = Q K^T : four 16x16 C tiles (64 keys), 4 K-chunks of 32 over D ----
    // Preload all 4 B fragments per kc into distinct registers so the 8
    // ds_load_b128 issue back-to-back and overlap the WMMA stream.
    v8f s[4];
#pragma unroll
    for (int nt = 0; nt < 4; ++nt)
#pragma unroll
      for (int r = 0; r < 8; ++r) s[nt][r] = 0.f;
#pragma unroll
    for (int kc = 0; kc < 4; ++kc) {
      v16bf bk[4];
#pragma unroll
      for (int nt = 0; nt < 4; ++nt)
        bk[nt] = *(const v16bf*)&sK[16 * nt + ln][kc * 32 + 16 * h];
#pragma unroll
      for (int nt = 0; nt < 4; ++nt)
        s[nt] = wmma_bf16(aQ[kc], bk[nt], s[nt]);
    }

    // ---- online softmax: row max via DPP butterfly; sums via ones-WMMA ----
    float corr[8];
#pragma unroll
    for (int r = 0; r < 8; ++r) {
      float t = fmaxf(fmaxf(s[0][r], s[1][r]), fmaxf(s[2][r], s[3][r]));
      t = rowmax16(t);
      float mn = fmaxf(rm[r], t);
      corr[r]  = exp2f(rm[r] - mn);
      rm[r]    = mn;
#pragma unroll
      for (int j = 0; j < 4; ++j) s[j][r] = exp2f(s[j][r] - mn);
    }

    // ---- re-layout P: C fragments -> LDS (bf16) -> two A fragments ----
#pragma unroll
    for (int r = 0; r < 8; ++r) {
#pragma unroll
      for (int j = 0; j < 4; ++j)
        sh.s.p[wave][r + 8 * h][16 * j + ln] = (__bf16)s[j][r];
    }
    asm volatile("s_wait_dscnt 0" ::: "memory");
    v8bf p0 = *(const v8bf*)&sh.s.p[wave][ln][     16 * h];
    v8bf p1 = *(const v8bf*)&sh.s.p[wave][ln][ 8 + 16 * h];
    v8bf p2 = *(const v8bf*)&sh.s.p[wave][ln][32 + 16 * h];
    v8bf p3 = *(const v8bf*)&sh.s.p[wave][ln][40 + 16 * h];
    v16bf aP0 = __builtin_shufflevector(p0, p1, 0, 1, 2, 3, 4, 5, 6, 7,
                                        8, 9, 10, 11, 12, 13, 14, 15);
    v16bf aP1 = __builtin_shufflevector(p2, p3, 0, 1, 2, 3, 4, 5, 6, 7,
                                        8, 9, 10, 11, 12, 13, 14, 15);

    // ---- L = corr*L + P·1 ; O[nt] = corr*O[nt] + P·V ----
    {
      v8f ls = Lacc;
#pragma unroll
      for (int r = 0; r < 8; ++r) ls[r] *= corr[r];
      ls   = wmma_bf16(aP0, ones, ls);
      Lacc = wmma_bf16(aP1, ones, ls);
    }
    // Two nt-tiles per step: 4 preloaded B fragments feed 4 WMMAs.
#pragma unroll
    for (int nt = 0; nt < 8; nt += 2) {
      v16bf bv00 = *(const v16bf*)&sh.s.vt[16 * nt + ln]       [     16 * h];
      v16bf bv01 = *(const v16bf*)&sh.s.vt[16 * nt + ln]       [32 + 16 * h];
      v16bf bv10 = *(const v16bf*)&sh.s.vt[16 * (nt + 1) + ln] [     16 * h];
      v16bf bv11 = *(const v16bf*)&sh.s.vt[16 * (nt + 1) + ln] [32 + 16 * h];
      v8f o0 = O[nt], o1 = O[nt + 1];
#pragma unroll
      for (int r = 0; r < 8; ++r) { o0[r] *= corr[r]; o1[r] *= corr[r]; }
      o0        = wmma_bf16(aP0, bv00, o0);
      o1        = wmma_bf16(aP0, bv10, o1);
      O[nt]     = wmma_bf16(aP1, bv01, o0);
      O[nt + 1] = wmma_bf16(aP1, bv11, o1);
    }
    __syncthreads();
  }

  // ---- combine across branch-wave pairs: out = O1/L1 - sigmoid(lam)*O2/L2 ----
  const float lam = 1.0f / (1.0f + __expf(-lamLogit[0]));
  float inv[8];
#pragma unroll
  for (int r = 0; r < 8; ++r) inv[r] = 1.0f / Lacc[r];

#pragma unroll
  for (int g = 0; g < 2; ++g) {     // two groups of 4 d-tiles (LDS reuse)
    if (branch == 1) {
#pragma unroll
      for (int j = 0; j < 4; ++j)
#pragma unroll
        for (int r = 0; r < 8; ++r)
          sh.combine[wv][r + 8 * h][j * 16 + ln] = lam * (O[4 * g + j][r] * inv[r]);
    }
    __syncthreads();
    if (branch == 0) {
#pragma unroll
      for (int j = 0; j < 4; ++j)
#pragma unroll
        for (int r = 0; r < 8; ++r) {
          int row = q0 + r + 8 * h;
          int col = (4 * g + j) * 16 + ln;
          out[((size_t)b * KQ + row) * DIM + col] =
              O[4 * g + j][r] * inv[r] - sh.combine[wv][r + 8 * h][j * 16 + ln];
        }
    }
    __syncthreads();
  }
}

extern "C" void kernel_launch(void* const* d_in, const int* in_sizes, int n_in,
                              void* d_out, int out_size, void* d_ws, size_t ws_size,
                              hipStream_t stream) {
  (void)in_sizes; (void)n_in; (void)out_size; (void)d_ws; (void)ws_size;
  const float* Q1  = (const float*)d_in[0];
  const float* Q2  = (const float*)d_in[1];
  const float* K1  = (const float*)d_in[2];
  const float* K2  = (const float*)d_in[3];
  const float* V   = (const float*)d_in[4];
  const float* lam = (const float*)d_in[5];

  dim3 grid(KQ / BQ, BATCH);   // (16, 16) blocks
  dim3 block(512);             // 16 wave32s: 8 q-tiles x 2 branches
  hipLaunchKernelGGL(diff_attn_kernel, grid, block, 0, stream,
                     Q1, Q2, K1, K2, V, lam, (float*)d_out);
}
